// CosineMixingLayer_28097676050518
// MI455X (gfx1250) — matrix operand order repaired and checked by reading, not compiled, not run on hardware
//
#include <hip/hip_runtime.h>
#include <hip/hip_fp16.h>

typedef __attribute__((ext_vector_type(16))) _Float16 v16h;
typedef __attribute__((ext_vector_type(8)))  _Float16 v8h;
typedef __attribute__((ext_vector_type(4)))  _Float16 v4h;
typedef __attribute__((ext_vector_type(8)))  float    v8f;

#define SEQ    4096
#define DMODEL 1024
#define NBATCH 8

#define BM 128
#define BN 128
#define BK 32
#define LDSS 40   // padded LDS row stride in f16 (80 B: 16B-aligned, conflict-free)

// ---------------------------------------------------------------------------
// CDNA5 async global->LDS copy (ISA 15.18.3 opcode 98), ASYNCcnt-tracked.
// LDS dest = per-lane 32-bit LDS byte offset (low 32 bits of generic pointer,
// per the aperture rule). GV mode: 64-bit vaddr, no saddr.
// ---------------------------------------------------------------------------
__device__ __forceinline__ void async_copy_b128(unsigned lds_off, const _Float16* g) {
    asm volatile("global_load_async_to_lds_b128 %0, %1, off"
                 :: "v"(lds_off), "v"(g)
                 : "memory");
}
__device__ __forceinline__ unsigned lds_off_u32(const void* p) {
    return (unsigned)(size_t)p;
}

// ---------------------------------------------------------------------------
// Kernel 1: orthonormal DCT-II matrix C[k][n] in f16.
// C[k][n] = s_k * cos(pi*k*(2n+1)/(2*SEQ)), s_0=1/64, s_k=sqrt(2)/64.
// Exact integer phase reduction mod 16384 (power of two) keeps cos args small.
// ---------------------------------------------------------------------------
__global__ void __launch_bounds__(256) build_dct_f16(_Float16* __restrict__ Cmat) {
    unsigned gid = blockIdx.x * 256u + threadIdx.x;
    unsigned k = gid >> 12;
    unsigned n = gid & 4095u;
    unsigned u = (k * (2u * n + 1u)) & 16383u;
    float ang = (float)u * 3.83495196971410e-4f;            // pi / 8192
    float s = (k == 0u) ? 0.015625f : 0.022097086912079608f;
    Cmat[gid] = (_Float16)(s * cosf(ang));
}

// ---------------------------------------------------------------------------
// Kernel 2: x f32 [b][k][d] -> xT f16 [b][d][k]  (tiled transpose + convert).
// Hoists the f32->f16 transpose out of the GEMM hot loop.
// ---------------------------------------------------------------------------
__global__ void __launch_bounds__(256) transpose_x_f16(const float* __restrict__ x,
                                                       _Float16* __restrict__ xT) {
    __shared__ _Float16 tile[32][40];                       // padded vs bank conflicts
    const unsigned kTiles = SEQ / 32, dTiles = DMODEL / 32;
    unsigned bid = blockIdx.x;
    unsigned b  = bid / (kTiles * dTiles);
    unsigned rr = bid % (kTiles * dTiles);
    unsigned k0 = (rr % kTiles) * 32u;
    unsigned d0 = (rr / kTiles) * 32u;
    const float* xb  = x  + (size_t)b * SEQ * DMODEL;
    _Float16*    xtb = xT + (size_t)b * DMODEL * SEQ;

    unsigned t  = threadIdx.x;
    unsigned dl = t & 31u, kg = t >> 5;                     // coalesced f32 reads
    #pragma unroll
    for (int j = 0; j < 4; ++j) {
        unsigned kl = kg * 4u + j;
        tile[kl][dl] = (_Float16)xb[(size_t)(k0 + kl) * DMODEL + d0 + dl];
    }
    __syncthreads();
    unsigned dl2 = t >> 3, kl0 = (t & 7u) * 4u;             // 8B vector writes
    v4h v;
    #pragma unroll
    for (int j = 0; j < 4; ++j) v[j] = tile[kl0 + j][dl2];
    *(v4h*)(xtb + (size_t)(d0 + dl2) * SEQ + k0 + kl0) = v;
}

// ---------------------------------------------------------------------------
// Kernel 3: out[b] = C * x[b] via WMMA. 256 thr = 8 waves (4 M x 2 N),
// 128x128 tile, BK=32, double-buffered LDS filled by async global->LDS b128.
// ---------------------------------------------------------------------------
union HFrag { v16h v; v8h h[2]; };

__device__ __forceinline__ void stage_async(const _Float16* Asrc, const _Float16* Bsrc,
                                            unsigned tid,
                                            const _Float16* bufA, const _Float16* bufB) {
    #pragma unroll
    for (int i = 0; i < 2; ++i) {
        unsigned idx = tid + 256u * i;                      // 0..511
        unsigned row = idx >> 2;                            // 0..127
        unsigned ch  = idx & 3u;                            // 4 x 8 f16 chunks
        async_copy_b128(lds_off_u32(bufA + row * LDSS + ch * 8u),
                        Asrc + (size_t)row * SEQ + ch * 8u);
        async_copy_b128(lds_off_u32(bufB + row * LDSS + ch * 8u),
                        Bsrc + (size_t)row * SEQ + ch * 8u);
    }
}

__device__ __forceinline__ void mma_step(const _Float16* __restrict__ lA,
                                         const _Float16* __restrict__ lB,
                                         unsigned waveM, unsigned waveN,
                                         unsigned lrow, unsigned half,
                                         v8f acc[2][4]) {
    HFrag a[2];
    #pragma unroll
    for (int mi = 0; mi < 2; ++mi) {
        const _Float16* ap = lA + (waveM * 32u + mi * 16u + lrow) * LDSS;
        a[mi].h[0] = *(const v8h*)(ap + half * 8u);         // K = half*8   + 0..7
        a[mi].h[1] = *(const v8h*)(ap + 16u + half * 8u);   // K = 16+half*8+ 0..7
    }
    #pragma unroll
    for (int ni = 0; ni < 4; ++ni) {
        HFrag bf;
        const _Float16* bp = lB + (waveN * 64u + ni * 16u + lrow) * LDSS;
        bf.h[0] = *(const v8h*)(bp + half * 8u);            // K = half*8   + 0..7
        bf.h[1] = *(const v8h*)(bp + 16u + half * 8u);      // K = 16+half*8+ 0..7
        #pragma unroll
        for (int mi = 0; mi < 2; ++mi) {
            acc[mi][ni] = __builtin_amdgcn_wmma_f32_16x16x32_f16(
                false, a[mi].v, false, bf.v, (short)0, acc[mi][ni], false, false);
        }
    }
}

__global__ void __launch_bounds__(256)
dct_gemm(const _Float16* __restrict__ Cmat,
         const _Float16* __restrict__ xT,
         float*          __restrict__ out) {
    __shared__ __align__(16) _Float16 ldsA[2][BM * LDSS];   // C tile  [m][k]
    __shared__ __align__(16) _Float16 ldsB[2][BN * LDSS];   // xT tile [n][k]

    const unsigned mTiles = SEQ / BM;                       // 32
    const unsigned nTiles = DMODEL / BN;                    // 8
    const unsigned bid = blockIdx.x;
    const unsigned b   = bid / (mTiles * nTiles);
    const unsigned rem = bid % (mTiles * nTiles);
    const unsigned m0  = (rem % mTiles) * BM;
    const unsigned n0  = (rem / mTiles) * BN;

    const unsigned tid   = threadIdx.x;
    const unsigned wave  = tid >> 5;
    const unsigned lane  = tid & 31u;
    const unsigned waveM = wave & 3u;
    const unsigned waveN = wave >> 2;
    const unsigned lrow  = lane & 15u;
    const unsigned half  = lane >> 4;

    const _Float16* Abase = Cmat + (size_t)m0 * SEQ;
    const _Float16* Bbase = xT + (size_t)b * DMODEL * SEQ + (size_t)n0 * SEQ;

    v8f acc[2][4];
    #pragma unroll
    for (int mi = 0; mi < 2; ++mi)
        #pragma unroll
        for (int ni = 0; ni < 4; ++ni)
            #pragma unroll
            for (int e = 0; e < 8; ++e)
                acc[mi][ni][e] = 0.0f;

    const unsigned nsteps = SEQ / BK;                       // 128
    stage_async(Abase, Bbase, tid, ldsA[0], ldsB[0]);       // prologue: tile 0

    for (unsigned t = 0; t < nsteps; ++t) {
        unsigned p = t & 1u;
        if (t + 1 < nsteps) {
            stage_async(Abase + (t + 1) * BK, Bbase + (t + 1) * BK,
                        tid, ldsA[p ^ 1u], ldsB[p ^ 1u]);
            // 4 async ops of tile t+1 outstanding; in-order => tile t resident
            asm volatile("s_wait_asynccnt 0x4" ::: "memory");
        } else {
            asm volatile("s_wait_asynccnt 0x0" ::: "memory");
        }
        __syncthreads();                                    // tile t visible to all
        mma_step(ldsA[p], ldsB[p], waveM, waveN, lrow, half, acc);
        __syncthreads();                                    // done reading buf p
    }

    // epilogue: C/D layout -> VGPR j holds M = j + 8*half, lane&15 = N
    float* ob = out + (size_t)b * SEQ * DMODEL;
    #pragma unroll
    for (int mi = 0; mi < 2; ++mi) {
        #pragma unroll
        for (int ni = 0; ni < 4; ++ni) {
            unsigned col   = n0 + waveN * 64u + ni * 16u + lrow;
            unsigned rbase = m0 + waveM * 32u + mi * 16u + half * 8u;
            #pragma unroll
            for (int j = 0; j < 8; ++j)
                ob[(size_t)(rbase + j) * DMODEL + col] = acc[mi][ni][j];
        }
    }
}

// ---------------------------------------------------------------------------
extern "C" void kernel_launch(void* const* d_in, const int* in_sizes, int n_in,
                              void* d_out, int out_size, void* d_ws, size_t ws_size,
                              hipStream_t stream) {
    const float* x = (const float*)d_in[0];
    float* out     = (float*)d_out;
    _Float16* Cmat = (_Float16*)d_ws;                              // 32 MiB
    _Float16* xT   = (_Float16*)((char*)d_ws + (size_t)SEQ * SEQ * 2); // 64 MiB

    build_dct_f16<<<(SEQ * SEQ) / 256, 256, 0, stream>>>(Cmat);
    transpose_x_f16<<<NBATCH * (SEQ / 32) * (DMODEL / 32), 256, 0, stream>>>(x, xT);
    dct_gemm<<<NBATCH * (SEQ / BM) * (DMODEL / BN), 256, 0, stream>>>(Cmat, xT, out);
}
